// DeepSeekMOE_52106543235531
// MI455X (gfx1250) — matrix operand order repaired and checked by reading, not compiled, not run on hardware
//
#include <hip/hip_runtime.h>
#include <hip/hip_bf16.h>
#include <math.h>

// ---- problem constants (match reference) ----
#define H_DIM 2048
#define I_DIM 1408
#define NE    7
#define T_TOK 8192           // B*S = 4*2048
#define KT1   (H_DIM / 32)   // 64 k-steps for gate/up GEMM
#define NT1   (I_DIM / 16)   // 88 n-tiles of I
#define KT2   (I_DIM / 32)   // 44 k-steps for down GEMM
#define NT2   (H_DIM / 16)   // 128 n-tiles of H
#define GM    4              // m-tiles (of 16) per wave
#define GN    2              // n-tiles (of 16) per wave

typedef __attribute__((ext_vector_type(16))) __bf16 v16bf;
typedef __attribute__((ext_vector_type(8)))  float  v8f;

union FragU {
  v16bf bf;
  uint4 u[2];
};

__device__ __forceinline__ unsigned short f2bf(float f) {
  unsigned int u = __float_as_uint(f);
  unsigned int r = u + 0x7fffu + ((u >> 16) & 1u);   // round-to-nearest-even
  return (unsigned short)(r >> 16);
}

// ---- workspace layout (bytes) ----
constexpr size_t OFF_XB   = 0;
constexpr size_t SZ_XB    = (size_t)T_TOK * H_DIM * 2;         // 33.5 MB bf16 x
constexpr size_t OFF_WF   = OFF_XB + SZ_XB;
constexpr size_t SZ_WF    = (size_t)8 * 3 * H_DIM * I_DIM * 2; // 138.4 MB bf16 weight fragments
constexpr size_t OFF_ACT  = OFF_WF + SZ_WF;
constexpr size_t SZ_ACT   = (size_t)T_TOK * I_DIM * 2;         // 23.1 MB act (reused per expert)
constexpr size_t OFF_PERM = OFF_ACT + SZ_ACT;
constexpr size_t SZ_PERM  = (size_t)NE * T_TOK * 4;
constexpr size_t OFF_WGT  = OFF_PERM + SZ_PERM;
constexpr size_t OFF_CNT  = OFF_WGT + SZ_PERM;

// -------------------------------------------------------------------------
__global__ void k_zero_cnt(int* cnt) {
  if (threadIdx.x < 8) cnt[threadIdx.x] = 0;
}

// fp32 -> bf16 row-major copy of x
__global__ void k_convert_x(const float* __restrict__ x, unsigned short* __restrict__ xb) {
  size_t i = ((size_t)blockIdx.x * blockDim.x + threadIdx.x) * 4;
  if (i >= (size_t)T_TOK * H_DIM) return;
  float4 v = *(const float4*)(x + i);
  ushort4 o;
  o.x = f2bf(v.x); o.y = f2bf(v.y); o.z = f2bf(v.z); o.w = f2bf(v.w);
  *(ushort4*)(xb + i) = o;
}

// Convert a row-major [Kdim x N] fp32 matrix into WMMA B-fragment order.
// B 32x16 bf16 frag in contiguous k order (A loads match): lane = h*16 + n, lane holds K = kt*32 + h*16 + j,
// j=0..15, stored as 32 contiguous bytes per lane.
__global__ void k_convert_frag(const float* __restrict__ src, unsigned short* __restrict__ dst,
                               int Kdim, int N) {
  int tid = blockIdx.x * blockDim.x + threadIdx.x;
  int ntiles = N >> 4;
  int total = (Kdim >> 5) * ntiles * 32;
  if (tid >= total) return;
  int lane = tid & 31, f = tid >> 5;
  int kt = f / ntiles, nt = f - kt * ntiles;
  int hh = lane >> 4, c = lane & 15;
  int col = nt * 16 + c;
  int kbase = kt * 32 + hh * 16;
  unsigned int o[8];
#pragma unroll
  for (int j = 0; j < 8; j++) {
    unsigned short lo = f2bf(src[(size_t)(kbase + 2 * j) * N + col]);
    unsigned short hi = f2bf(src[(size_t)(kbase + 2 * j + 1) * N + col]);
    o[j] = (unsigned)lo | ((unsigned)hi << 16);
  }
  uint4* d = (uint4*)(dst + (size_t)tid * 16);
  d[0] = make_uint4(o[0], o[1], o[2], o[3]);
  d[1] = make_uint4(o[4], o[5], o[6], o[7]);
}

// One wave per token: logits = x @ w_router + bias, sigmoid, top-2, renorm,
// scatter token into per-expert lists.
__global__ void k_router(const float* __restrict__ x, const float* __restrict__ wrt,
                         const float* __restrict__ bias,
                         int* __restrict__ cnt, int* __restrict__ perm, float* __restrict__ wgt) {
  int t = blockIdx.x, lane = threadIdx.x;
  const float* xr = x + (size_t)t * H_DIM;
  float acc[NE];
#pragma unroll
  for (int e = 0; e < NE; e++) acc[e] = 0.f;
  for (int hh = lane; hh < H_DIM; hh += 32) {
    float xv = xr[hh];
    const float* w = wrt + (size_t)hh * NE;
#pragma unroll
    for (int e = 0; e < NE; e++) acc[e] += xv * w[e];
  }
#pragma unroll
  for (int e = 0; e < NE; e++) {
#pragma unroll
    for (int off = 16; off > 0; off >>= 1) acc[e] += __shfl_xor(acc[e], off, 32);
  }
  if (lane == 0) {
    float p[NE];
#pragma unroll
    for (int e = 0; e < NE; e++) p[e] = 1.f / (1.f + __expf(-(acc[e] + bias[e])));
    int e0 = 0;
#pragma unroll
    for (int e = 1; e < NE; e++) if (p[e] > p[e0]) e0 = e;
    int e1 = -1;
#pragma unroll
    for (int e = 0; e < NE; e++) if (e != e0 && (e1 < 0 || p[e] > p[e1])) e1 = e;
    float s = p[e0] + p[e1];
    float s0 = p[e0] / s, s1 = p[e1] / s;
    int p0 = atomicAdd(&cnt[e0], 1); perm[e0 * T_TOK + p0] = t; wgt[e0 * T_TOK + p0] = s0;
    int p1 = atomicAdd(&cnt[e1], 1); perm[e1 * T_TOK + p1] = t; wgt[e1 * T_TOK + p1] = s1;
  }
}

// GEMM1: act = silu(x@wg) * (x@wu) for this expert's token list.
// One wave -> 64(M) x 32(N) macro-tile, gate+up fused:
// per k-step: 4 A-frag loads + 4 B-frag loads (gate+up) -> 16 WMMAs.
__global__ void k_gemm1(const unsigned short* __restrict__ xb,
                        const unsigned short* __restrict__ wg,
                        const unsigned short* __restrict__ wu,
                        const int* __restrict__ perm, const int* __restrict__ cnt,
                        unsigned short* __restrict__ act) {
  int ntb = blockIdx.x * GN;          // first n-tile (of 16)
  int mb  = blockIdx.y * (GM * 16);   // first row
  int lane = threadIdx.x;
  int count = cnt ? *cnt : T_TOK;
  if (mb >= count) return;
  int hh = lane >> 4, c = lane & 15;

  const unsigned short* arow[GM];
#pragma unroll
  for (int mi = 0; mi < GM; mi++) {
    int pos = mb + mi * 16 + c;
    int tok = perm ? (pos < count ? perm[pos] : 0) : pos;
    arow[mi] = xb + (size_t)tok * H_DIM;
  }

  v8f zero = {};
  v8f cg[GM][GN], cu[GM][GN];
#pragma unroll
  for (int mi = 0; mi < GM; mi++)
#pragma unroll
    for (int ni = 0; ni < GN; ni++) { cg[mi][ni] = zero; cu[mi][ni] = zero; }

  for (int kt = 0; kt < KT1; ++kt) {
    FragU a[GM], bg[GN], bu[GN];
#pragma unroll
    for (int mi = 0; mi < GM; mi++) {
      // A frag, contiguous k order (matches B frags): lane(h,m) holds K = kt*32 + h*16 + 0..15
      const unsigned short* ab = arow[mi] + kt * 32 + hh * 16;
      a[mi].u[0] = *(const uint4*)(ab);
      a[mi].u[1] = *(const uint4*)(ab + 8);
    }
#pragma unroll
    for (int ni = 0; ni < GN; ni++) {
      size_t fb = (((size_t)kt * NT1 + ntb + ni) * 32 + lane) * 16;
      bg[ni].u[0] = *(const uint4*)(wg + fb); bg[ni].u[1] = *(const uint4*)(wg + fb + 8);
      bu[ni].u[0] = *(const uint4*)(wu + fb); bu[ni].u[1] = *(const uint4*)(wu + fb + 8);
    }
#pragma unroll
    for (int mi = 0; mi < GM; mi++)
#pragma unroll
      for (int ni = 0; ni < GN; ni++) {
        cg[mi][ni] = __builtin_amdgcn_wmma_f32_16x16x32_bf16(
            false, a[mi].bf, false, bg[ni].bf, (short)0, cg[mi][ni], false, false);
        cu[mi][ni] = __builtin_amdgcn_wmma_f32_16x16x32_bf16(
            false, a[mi].bf, false, bu[ni].bf, (short)0, cu[mi][ni], false, false);
      }
  }

#pragma unroll
  for (int mi = 0; mi < GM; mi++)
#pragma unroll
    for (int ni = 0; ni < GN; ni++) {
      int col = (ntb + ni) * 16 + c;
#pragma unroll
      for (int r = 0; r < 8; r++) {   // C layout: VGPR r -> M = r + 8*h, N = c
        int m = mb + mi * 16 + r + 8 * hh;
        float g = cg[mi][ni][r], u = cu[mi][ni][r];
        float v = (g / (1.f + __expf(-g))) * u;   // silu(g) * u
        act[(size_t)m * I_DIM + col] = f2bf(v);
      }
    }
}

// GEMM2: out[tok] (+)= (act @ wd) * weight. One wave -> 64(M) x 32(N).
// Shared expert (perm==null) plain-stores; routed experts accumulate
// (stream-serialized launches => no float atomics, deterministic).
__global__ void k_gemm2(const unsigned short* __restrict__ act,
                        const unsigned short* __restrict__ wd,
                        const int* __restrict__ perm, const float* __restrict__ wgt,
                        const int* __restrict__ cnt, float* __restrict__ out) {
  int ntb = blockIdx.x * GN;
  int mb  = blockIdx.y * (GM * 16);
  int lane = threadIdx.x;
  int count = cnt ? *cnt : T_TOK;
  if (mb >= count) return;
  int hh = lane >> 4, c = lane & 15;

  const unsigned short* arow[GM];
#pragma unroll
  for (int mi = 0; mi < GM; mi++)
    arow[mi] = act + (size_t)(mb + mi * 16 + c) * I_DIM;

  v8f zero = {};
  v8f acc[GM][GN];
#pragma unroll
  for (int mi = 0; mi < GM; mi++)
#pragma unroll
    for (int ni = 0; ni < GN; ni++) acc[mi][ni] = zero;

  for (int kt = 0; kt < KT2; ++kt) {
    FragU a[GM], b[GN];
#pragma unroll
    for (int mi = 0; mi < GM; mi++) {
      const unsigned short* ab = arow[mi] + kt * 32 + hh * 16;
      a[mi].u[0] = *(const uint4*)(ab);
      a[mi].u[1] = *(const uint4*)(ab + 8);
    }
#pragma unroll
    for (int ni = 0; ni < GN; ni++) {
      size_t fb = (((size_t)kt * NT2 + ntb + ni) * 32 + lane) * 16;
      b[ni].u[0] = *(const uint4*)(wd + fb); b[ni].u[1] = *(const uint4*)(wd + fb + 8);
    }
#pragma unroll
    for (int mi = 0; mi < GM; mi++)
#pragma unroll
      for (int ni = 0; ni < GN; ni++)
        acc[mi][ni] = __builtin_amdgcn_wmma_f32_16x16x32_bf16(
            false, a[mi].bf, false, b[ni].bf, (short)0, acc[mi][ni], false, false);
  }

  if (perm) {
#pragma unroll
    for (int mi = 0; mi < GM; mi++)
#pragma unroll
      for (int r = 0; r < 8; r++) {
        int row = mb + mi * 16 + r + 8 * hh;
        if (row < count) {
          int tok = perm[row];
          float w = wgt[row];
          float* o = out + (size_t)tok * H_DIM;
#pragma unroll
          for (int ni = 0; ni < GN; ni++) {
            int col = (ntb + ni) * 16 + c;
            o[col] += acc[mi][ni][r] * w;
          }
        }
      }
  } else {
#pragma unroll
    for (int mi = 0; mi < GM; mi++)
#pragma unroll
      for (int r = 0; r < 8; r++) {
        int row = mb + mi * 16 + r + 8 * hh;
        float* o = out + (size_t)row * H_DIM;
#pragma unroll
        for (int ni = 0; ni < GN; ni++) {
          int col = (ntb + ni) * 16 + c;
          o[col] = acc[mi][ni][r];
        }
      }
  }
}

// -------------------------------------------------------------------------
extern "C" void kernel_launch(void* const* d_in, const int* in_sizes, int n_in,
                              void* d_out, int out_size, void* d_ws, size_t ws_size,
                              hipStream_t stream) {
  (void)in_sizes; (void)n_in; (void)out_size; (void)ws_size;
  const float* x        = (const float*)d_in[0];
  const float* ws_gate  = (const float*)d_in[1];
  const float* ws_up    = (const float*)d_in[2];
  const float* ws_down  = (const float*)d_in[3];
  const float* wr_gate  = (const float*)d_in[4];
  const float* wr_up    = (const float*)d_in[5];
  const float* wr_down  = (const float*)d_in[6];
  const float* w_router = (const float*)d_in[7];
  const float* r_bias   = (const float*)d_in[8];
  float* out = (float*)d_out;

  char* ws = (char*)d_ws;
  unsigned short* xb  = (unsigned short*)(ws + OFF_XB);
  unsigned short* wf  = (unsigned short*)(ws + OFF_WF);
  unsigned short* act = (unsigned short*)(ws + OFF_ACT);
  int*   perm = (int*)(ws + OFF_PERM);
  float* wgt  = (float*)(ws + OFF_WGT);
  int*   cnt  = (int*)(ws + OFF_CNT);

  k_zero_cnt<<<1, 32, 0, stream>>>(cnt);

  { size_t n = (size_t)T_TOK * H_DIM / 4;
    k_convert_x<<<(unsigned)((n + 255) / 256), 256, 0, stream>>>(x, xb); }

  const size_t MAT = (size_t)H_DIM * I_DIM;       // elements per weight matrix
  int totGU = (H_DIM / 32) * (I_DIM / 16) * 32;   // lane-units for gate/up
  int totD  = (I_DIM / 32) * (H_DIM / 16) * 32;   // lane-units for down
  for (int s = 0; s < 8; s++) {                   // slots 0..6 routed, 7 shared
    const float* g = (s < 7) ? wr_gate + (size_t)s * MAT : ws_gate;
    const float* u = (s < 7) ? wr_up   + (size_t)s * MAT : ws_up;
    const float* d = (s < 7) ? wr_down + (size_t)s * MAT : ws_down;
    unsigned short* base = wf + (size_t)s * 3 * MAT;
    k_convert_frag<<<(totGU + 255) / 256, 256, 0, stream>>>(g, base,           H_DIM, I_DIM);
    k_convert_frag<<<(totGU + 255) / 256, 256, 0, stream>>>(u, base + MAT,     H_DIM, I_DIM);
    k_convert_frag<<<(totD  + 255) / 256, 256, 0, stream>>>(d, base + 2 * MAT, I_DIM, H_DIM);
  }

  k_router<<<T_TOK, 32, 0, stream>>>(x, w_router, r_bias, cnt, perm, wgt);

  dim3 g1(NT1 / GN, T_TOK / (GM * 16));   // (44, 128)
  dim3 g2(NT2 / GN, T_TOK / (GM * 16));   // (64, 128)

  // Shared expert first: plain stores initialize out.
  {
    unsigned short* base = wf + (size_t)7 * 3 * MAT;
    k_gemm1<<<g1, 32, 0, stream>>>(xb, base, base + MAT, nullptr, nullptr, act);
    k_gemm2<<<g2, 32, 0, stream>>>(act, base + 2 * MAT, nullptr, nullptr, nullptr, out);
  }
  // Routed experts: stream-serialized accumulation into out.
  for (int e = 0; e < NE; e++) {
    unsigned short* base = wf + (size_t)e * 3 * MAT;
    k_gemm1<<<g1, 32, 0, stream>>>(xb, base, base + MAT, perm + e * T_TOK, cnt + e, act);
    k_gemm2<<<g2, 32, 0, stream>>>(act, base + 2 * MAT, perm + e * T_TOK, wgt + e * T_TOK, cnt + e, out);
  }
}
